// BiLSTM_CRF_61735859913511
// MI455X (gfx1250) — matrix operand order repaired and checked by reading, not compiled, not run on hardware
//
#include <hip/hip_runtime.h>
#include <stdint.h>

typedef __attribute__((ext_vector_type(16))) __bf16 v16bf;
typedef __attribute__((ext_vector_type(8)))  __bf16 v8bf;
typedef __attribute__((ext_vector_type(8)))  float  v8f;

#define B_    64
#define T_    512
#define E_    512
#define H_    512
#define FOURH 2048
#define KTAGS 23
#define START_TAG 1
#define END_TAG   2
#define NEGV  (-10000.0f)
#define HSTR  520   // LDS row stride (elems): 260 dwords == 4 banks mod 64 -> conflict-free frag reads

__device__ __forceinline__ uint16_t f2bf(float f) {
  union { float f; uint32_t u; } v; v.f = f;
  uint32_t u = v.u;
  uint32_t r = u + 0x7FFFu + ((u >> 16) & 1u);   // round-to-nearest-even
  return (uint16_t)(r >> 16);
}
__device__ __forceinline__ float bf2f(uint16_t b) {
  union { uint32_t u; float f; } v; v.u = ((uint32_t)b) << 16; return v.f;
}
__device__ __forceinline__ float sigmoidf_(float x) { return 1.0f / (1.0f + __expf(-x)); }

// A-fragment (16x32 bf16, MxK): lane = row (l15), two 8-elem K chunks:
//   elems 0..7  = K[16h .. 16h+7],  elems 8..15 = K[16h+8 .. 16h+15]
__device__ __forceinline__ v16bf load_a_frag(const uint16_t* rowk, int hh) {
  v8bf lo = *(const v8bf*)(rowk + 16 * hh);
  v8bf hi = *(const v8bf*)(rowk + 16 * hh + 8);
  return __builtin_shufflevector(lo, hi, 0,1,2,3,4,5,6,7,8,9,10,11,12,13,14,15);
}
// B-fragment (32x16 bf16, KxN): lane = col (l15), contiguous K[16h .. 16h+15]
__device__ __forceinline__ v16bf load_b_frag(const uint16_t* colk, int hh) {
  return *(const v16bf*)(colk + 16 * hh);
}

// ---------------------------------------------------------------- f32 -> bf16
__global__ void cvt_bf16_kernel(const float* __restrict__ src,
                                uint16_t* __restrict__ dst, int n) {
  int i = blockIdx.x * blockDim.x + threadIdx.x;
  int stride = gridDim.x * blockDim.x;
  for (; i < n; i += stride) dst[i] = f2bf(src[i]);
}

// --------------------------------------------- Xproj = X @ Wih^T + b  (bf16 out)
// M=32768, N=2048, K=512. Block 256 thr (8 waves), block tile 128M x 64N,
// wave tile 32x32 = 2x2 wmma frags.
__global__ __launch_bounds__(256)
void xproj_gemm_kernel(const uint16_t* __restrict__ X,   // [32768,512] bf16
                       const uint16_t* __restrict__ W,   // [2048,512]  bf16 (row n = B col n)
                       const float* __restrict__ bias,   // [2048]
                       uint16_t* __restrict__ out)       // [32768,2048] bf16
{
  const int lane = threadIdx.x & 31;
  const int wid  = threadIdx.x >> 5;
  const int l15  = lane & 15;
  const int hh   = lane >> 4;
  const int mbase = blockIdx.y * 128 + (wid >> 1) * 32;
  const int nbase = blockIdx.x * 64  + (wid & 1) * 32;

  const uint16_t* a0 = X + (size_t)(mbase +      l15) * E_;
  const uint16_t* a1 = X + (size_t)(mbase + 16 + l15) * E_;
  const uint16_t* b0 = W + (size_t)(nbase +      l15) * E_;
  const uint16_t* b1 = W + (size_t)(nbase + 16 + l15) * E_;

  v8f acc[2][2] = {};
  for (int kt = 0; kt < E_ / 32; ++kt) {
    const int ko = kt * 32;
    __builtin_prefetch(a0 + ko + 64, 0, 1);    // global_prefetch_b8, next-next K tile
    v16bf af0 = load_a_frag(a0 + ko, hh);
    v16bf af1 = load_a_frag(a1 + ko, hh);
    v16bf bf0 = load_b_frag(b0 + ko, hh);
    v16bf bf1 = load_b_frag(b1 + ko, hh);
    acc[0][0] = __builtin_amdgcn_wmma_f32_16x16x32_bf16(false, af0, false, bf0, (short)0, acc[0][0], false, false);
    acc[0][1] = __builtin_amdgcn_wmma_f32_16x16x32_bf16(false, af0, false, bf1, (short)0, acc[0][1], false, false);
    acc[1][0] = __builtin_amdgcn_wmma_f32_16x16x32_bf16(false, af1, false, bf0, (short)0, acc[1][0], false, false);
    acc[1][1] = __builtin_amdgcn_wmma_f32_16x16x32_bf16(false, af1, false, bf1, (short)0, acc[1][1], false, false);
  }
  #pragma unroll
  for (int mi = 0; mi < 2; ++mi)
    #pragma unroll
    for (int ni = 0; ni < 2; ++ni) {
      const int n = nbase + ni * 16 + l15;
      const float bn = bias[n];
      #pragma unroll
      for (int r = 0; r < 8; ++r) {
        const int m = mbase + mi * 16 + hh * 8 + r;   // C layout: VGPR r -> M = r + 8h
        out[(size_t)m * FOURH + n] = f2bf(acc[mi][ni][r] + bn);
      }
    }
}

// --------------------------------------------- persistent recurrent BiLSTM
// grid = 2 blocks (fwd/bwd), 1024 threads = 32 waves. h (bf16) lives in LDS,
// c lives in registers. Wave w owns hidden cols [16w,16w+16) for ALL 4 gates,
// so the LSTM cell update is fully register-local.
__global__ __launch_bounds__(1024)
void bilstm_recurrent_kernel(const uint16_t* __restrict__ Xf,   // [B*T,2048] bf16 (x@Wih_f^T+b_f)
                             const uint16_t* __restrict__ Xb,
                             const uint16_t* __restrict__ Whh_f, // [2048,512] bf16
                             const uint16_t* __restrict__ Whh_b,
                             float* __restrict__ hf_out,         // [B,T,512] f32
                             float* __restrict__ hb_out)
{
  extern __shared__ uint16_t h_lds[];                // [64][HSTR] bf16
  const bool rev = (blockIdx.x == 1);
  const uint16_t* Xp  = rev ? Xb    : Xf;
  const uint16_t* Whh = rev ? Whh_b : Whh_f;
  float* hout = rev ? hb_out : hf_out;

  const int lane = threadIdx.x & 31;
  const int w    = threadIdx.x >> 5;                 // 0..31
  const int l15  = lane & 15;
  const int hh   = lane >> 4;

  for (int i = threadIdx.x; i < 64 * HSTR; i += 1024) h_lds[i] = 0;
  float c[4][8];
  #pragma unroll
  for (int mt = 0; mt < 4; ++mt)
    #pragma unroll
    for (int r = 0; r < 8; ++r) c[mt][r] = 0.0f;
  __syncthreads();

  // B-operand row pointers: gate g, hidden-col block w  (Whh is B^T row-major)
  const uint16_t* wrow[4];
  #pragma unroll
  for (int g = 0; g < 4; ++g)
    wrow[g] = Whh + (size_t)(g * H_ + w * 16 + l15) * H_ + 16 * hh;

  const int j = w * 16 + l15;                        // this lane's hidden column

  for (int t = 0; t < T_; ++t) {
    const int tt = rev ? (T_ - 1 - t) : t;
    v8f acc[4][4] = {};                              // [gate][mTile]
    for (int kt = 0; kt < H_ / 32; ++kt) {
      const int ko = kt * 32;
      v16bf aF[4];
      #pragma unroll
      for (int mt = 0; mt < 4; ++mt)
        aF[mt] = load_a_frag(&h_lds[(mt * 16 + l15) * HSTR + ko], hh);
      #pragma unroll
      for (int g = 0; g < 4; ++g) {
        v16bf bF = *(const v16bf*)(wrow[g] + ko);
        #pragma unroll
        for (int mt = 0; mt < 4; ++mt)
          acc[g][mt] = __builtin_amdgcn_wmma_f32_16x16x32_bf16(
              false, aF[mt], false, bF, (short)0, acc[g][mt], false, false);
      }
    }
    __syncthreads();                                 // all reads of h(t-1) done

    #pragma unroll
    for (int mt = 0; mt < 4; ++mt) {
      #pragma unroll
      for (int r = 0; r < 8; ++r) {
        const int b = mt * 16 + hh * 8 + r;          // batch row (C layout)
        const uint16_t* xr = Xp + (size_t)(b * T_ + tt) * FOURH;
        const float zi = acc[0][mt][r] + bf2f(xr[0 * H_ + j]);
        const float zf = acc[1][mt][r] + bf2f(xr[1 * H_ + j]);
        const float zg = acc[2][mt][r] + bf2f(xr[2 * H_ + j]);
        const float zo = acc[3][mt][r] + bf2f(xr[3 * H_ + j]);
        const float ig = sigmoidf_(zi);
        const float fg = sigmoidf_(zf);
        const float gg = tanhf(zg);
        const float og = sigmoidf_(zo);
        const float cn = fg * c[mt][r] + ig * gg;
        c[mt][r] = cn;
        const float hn = og * tanhf(cn);
        hout[(size_t)(b * T_ + tt) * H_ + j] = hn;
        h_lds[b * HSTR + j] = f2bf(hn);
      }
    }
    __syncthreads();                                 // h(t) visible for next step
  }
}

// --------------------------------------------- feats = [hf|hb] @ W_h2t^T + b
// one wave per (b,t); lanes parallel over hidden dim, butterfly reduce.
__global__ __launch_bounds__(256)
void feats_kernel(const float* __restrict__ hf, const float* __restrict__ hb,
                  const float* __restrict__ Wt,   // [23,1024]
                  const float* __restrict__ bt,   // [23]
                  float* __restrict__ feats)      // [B*T,23]
{
  const int gw   = (blockIdx.x * blockDim.x + threadIdx.x) >> 5;
  const int lane = threadIdx.x & 31;
  if (gw >= B_ * T_) return;
  const float* hfr = hf + (size_t)gw * H_;
  const float* hbr = hb + (size_t)gw * H_;
  for (int k = 0; k < KTAGS; ++k) {
    const float* wk = Wt + (size_t)k * (2 * H_);
    float s = 0.0f;
    for (int j2 = lane; j2 < H_; j2 += 32)
      s += hfr[j2] * wk[j2] + hbr[j2] * wk[H_ + j2];
    #pragma unroll
    for (int off = 16; off > 0; off >>= 1) s += __shfl_xor(s, off, 32);
    if (lane == 0) feats[(size_t)gw * KTAGS + k] = s + bt[k];
  }
}

// --------------------------------------------- Viterbi: one wave per batch elem
__global__ __launch_bounds__(32)
void viterbi_kernel(const float* __restrict__ feats,    // [B,T,K]
                    const int* __restrict__ seq_len,    // [B]
                    const float* __restrict__ trans,    // [K,K] trans[next][prev]
                    int* __restrict__ bp_ws,            // [B,T,K]
                    float* __restrict__ out)            // [B] scores ++ [B,T] paths
{
  __shared__ float tr[KTAGS * KTAGS];
  __shared__ float fv[KTAGS];
  const int b = blockIdx.x;
  const int lane = threadIdx.x;
  for (int i = lane; i < KTAGS * KTAGS; i += 32) tr[i] = trans[i];
  if (lane < KTAGS) fv[lane] = (lane == START_TAG) ? 0.0f : NEGV;
  __syncthreads();
  const int L = seq_len[b];
  for (int t = 0; t < T_; ++t) {
    float newv = 0.0f; int bpv = lane;
    if (lane < KTAGS) {
      float best = -3.4e38f; int bi = 0;
      for (int p = 0; p < KTAGS; ++p) {
        const float v = fv[p] + tr[lane * KTAGS + p];
        if (v > best) { best = v; bi = p; }            // first-max, like argmax
      }
      const float ft = feats[(size_t)(b * T_ + t) * KTAGS + lane];
      const bool m = (t < L);
      newv = m ? (best + ft) : fv[lane];
      bpv  = m ? bi : lane;                            // identity bp when padded
      bp_ws[(size_t)(b * T_ + t) * KTAGS + lane] = bpv;
    }
    __syncthreads();
    if (lane < KTAGS) fv[lane] = newv;
    __syncthreads();
  }
  float v  = (lane < KTAGS) ? (fv[lane] + tr[END_TAG * KTAGS + lane]) : -3.4e38f;
  int  idx = (lane < KTAGS) ? lane : KTAGS;
  #pragma unroll
  for (int off = 16; off > 0; off >>= 1) {
    const float ov = __shfl_xor(v, off, 32);
    const int   oi = __shfl_xor(idx, off, 32);
    if (ov > v || (ov == v && oi < idx)) { v = ov; idx = oi; }
  }
  if (lane == 0) {
    out[b] = v;
    float* paths = out + B_;
    int tag = idx;
    paths[(size_t)b * T_ + (T_ - 1)] = (float)tag;
    for (int t = T_ - 2; t >= 0; --t) {
      tag = bp_ws[(size_t)(b * T_ + t + 1) * KTAGS + tag];
      paths[(size_t)b * T_ + t] = (float)tag;
    }
  }
}

// ----------------------------------------------------------------------------
extern "C" void kernel_launch(void* const* d_in, const int* in_sizes, int n_in,
                              void* d_out, int out_size, void* d_ws, size_t ws_size,
                              hipStream_t stream) {
  const float* texts   = (const float*)d_in[0];
  const int*   seq_len = (const int*)  d_in[1];
  const float* Wih_f   = (const float*)d_in[2];
  const float* Whh_f   = (const float*)d_in[3];
  const float* b_f     = (const float*)d_in[4];
  const float* Wih_b   = (const float*)d_in[5];
  const float* Whh_b   = (const float*)d_in[6];
  const float* b_b     = (const float*)d_in[7];
  const float* W_h2t   = (const float*)d_in[8];
  const float* b_h2t   = (const float*)d_in[9];
  const float* trans   = (const float*)d_in[10];
  float* out = (float*)d_out;

  char* ws = (char*)d_ws;
  size_t off = 0;
  auto alloc = [&](size_t bytes) {
    void* p = ws + off;
    off = (off + bytes + 255) & ~(size_t)255;
    return p;
  };
  uint16_t* texts_bf = (uint16_t*)alloc((size_t)B_ * T_ * E_ * 2);
  uint16_t* Wih_f_bf = (uint16_t*)alloc((size_t)FOURH * E_ * 2);
  uint16_t* Whh_f_bf = (uint16_t*)alloc((size_t)FOURH * H_ * 2);
  uint16_t* Wih_b_bf = (uint16_t*)alloc((size_t)FOURH * E_ * 2);
  uint16_t* Whh_b_bf = (uint16_t*)alloc((size_t)FOURH * H_ * 2);
  uint16_t* Xf       = (uint16_t*)alloc((size_t)B_ * T_ * FOURH * 2);
  uint16_t* Xb       = (uint16_t*)alloc((size_t)B_ * T_ * FOURH * 2);
  float*    hf       = (float*)   alloc((size_t)B_ * T_ * H_ * 4);
  float*    hb       = (float*)   alloc((size_t)B_ * T_ * H_ * 4);
  float*    feats    = (float*)   alloc((size_t)B_ * T_ * KTAGS * 4);
  int*      bp       = (int*)     alloc((size_t)B_ * T_ * KTAGS * 4);
  (void)ws_size; (void)in_sizes; (void)n_in; (void)out_size;

  // 1) precision conversion (bf16 operands for WMMA)
  cvt_bf16_kernel<<<4096, 256, 0, stream>>>(texts, texts_bf, B_ * T_ * E_);
  cvt_bf16_kernel<<<512,  256, 0, stream>>>(Wih_f, Wih_f_bf, FOURH * E_);
  cvt_bf16_kernel<<<512,  256, 0, stream>>>(Whh_f, Whh_f_bf, FOURH * H_);
  cvt_bf16_kernel<<<512,  256, 0, stream>>>(Wih_b, Wih_b_bf, FOURH * E_);
  cvt_bf16_kernel<<<512,  256, 0, stream>>>(Whh_b, Whh_b_bf, FOURH * H_);

  // 2) hoisted input projections: one big WMMA GEMM per direction
  dim3 gg(FOURH / 64, (B_ * T_) / 128);
  xproj_gemm_kernel<<<gg, 256, 0, stream>>>(texts_bf, Wih_f_bf, b_f, Xf);
  xproj_gemm_kernel<<<gg, 256, 0, stream>>>(texts_bf, Wih_b_bf, b_b, Xb);

  // 3) persistent recurrent kernel: fwd + bwd concurrently, h in LDS
  bilstm_recurrent_kernel<<<2, 1024, 64 * HSTR * 2, stream>>>(
      Xf, Xb, Whh_f_bf, Whh_b_bf, hf, hb);

  // 4) emissions
  feats_kernel<<<(B_ * T_ * 32) / 256, 256, 0, stream>>>(hf, hb, W_h2t, b_h2t, feats);

  // 5) Viterbi decode
  viterbi_kernel<<<B_, 32, 0, stream>>>(feats, seq_len, trans, bp, out);
}
